// VectorQuantizer_6021544149259
// MI455X (gfx1250) — matrix operand order repaired and checked by reading, not compiled, not run on hardware
//
#include <hip/hip_runtime.h>

typedef _Float16 v16h __attribute__((ext_vector_type(16)));
typedef _Float16 v8h  __attribute__((ext_vector_type(8)));
typedef float    v8f  __attribute__((ext_vector_type(8)));

#define F_   16
#define N_   8192
#define D_   64
#define K_   512
#define ROWS_PER_WG 128   // 8 waves x 16 rows
#define KH_COLS     256   // K columns staged per LDS phase
#define BROW        72    // padded LDS row (halves): 144B, 16B aligned, conflict-free reads

// ---------------------------------------------------------------------------
// Kernel 1: precompute ||w_k||^2 once per (f,k) -> d_ws (saves a full
// redundant pass over w in every main block).
// ---------------------------------------------------------------------------
__global__ __launch_bounds__(256) void vq_wnorm(const float* __restrict__ w,
                                                float* __restrict__ wnorm_g)
{
    const int k  = blockIdx.x * 256 + threadIdx.x;   // 0 .. F*K-1
    const int f  = k >> 9;
    const int kk = k & (K_ - 1);
    const float* wf = w + (size_t)f * D_ * K_;
    float s = 0.f;
    #pragma unroll 8
    for (int d = 0; d < D_; ++d) {
        float v = wf[d * K_ + kk];                   // coalesced over k
        s += v * v;
    }
    wnorm_g[k] = s;
}

// ---------------------------------------------------------------------------
// Kernel 2: fused f16-WMMA GEMM (x . w) -> distance argmin -> exact f32 code
// gather + output write + deterministic per-block partial loss.
// One block = 8 waves, 128 rows of N, one feature f. Grid = 16*64 = 1024.
// ---------------------------------------------------------------------------
__global__ __launch_bounds__(256) void vq_main(const float* __restrict__ x,
                                               const float* __restrict__ w,
                                               const float* __restrict__ wnorm_g,
                                               float* __restrict__ out,
                                               float* __restrict__ partial)
{
    __shared__ __attribute__((aligned(16))) _Float16 Bs[KH_COLS * BROW]; // 36,864 B
    __shared__ float wn[K_];                                             //  2,048 B
    __shared__ float wls[8];

    const int tid  = threadIdx.x;
    const int lane = tid & 31;
    const int wv   = tid >> 5;
    const int bid  = blockIdx.x;
    const int f    = bid >> 6;        // 64 row-blocks per feature
    const int rb   = bid & 63;
    const int n0   = rb * ROWS_PER_WG + wv * 16;

    const int m  = lane & 15;         // A row / B column within tile
    const int hi = (lane >> 4) & 1;   // half-wave selector

    const float* xf = x + (size_t)f * N_ * D_;
    const float* wf = w + (size_t)f * D_ * K_;

    // ---- stage precomputed ||w_k||^2 to LDS (2 coalesced loads/thread) ----
    for (int k = tid; k < K_; k += 256) wn[k] = wnorm_g[f * K_ + k];

    // ---- load A tile (16 rows x 64 d) into two v16h (ISA 16-bit A layout):
    //      lane m=row; slot j -> d = c*32 + hi*8 + (j<8 ? j : j+8) ----
    const float* xr = xf + (size_t)(n0 + m) * D_;
    v16h a0, a1;
    #pragma unroll
    for (int c = 0; c < 2; ++c) {
        const int dbase = c * 32 + hi * 8;
        float4 r0 = *(const float4*)(xr + dbase);
        float4 r1 = *(const float4*)(xr + dbase + 4);
        float4 r2 = *(const float4*)(xr + dbase + 16);
        float4 r3 = *(const float4*)(xr + dbase + 20);
        v16h& a = c ? a1 : a0;
        a[0]  = (_Float16)r0.x; a[1]  = (_Float16)r0.y;
        a[2]  = (_Float16)r0.z; a[3]  = (_Float16)r0.w;
        a[4]  = (_Float16)r1.x; a[5]  = (_Float16)r1.y;
        a[6]  = (_Float16)r1.z; a[7]  = (_Float16)r1.w;
        a[8]  = (_Float16)r2.x; a[9]  = (_Float16)r2.y;
        a[10] = (_Float16)r2.z; a[11] = (_Float16)r2.w;
        a[12] = (_Float16)r3.x; a[13] = (_Float16)r3.y;
        a[14] = (_Float16)r3.z; a[15] = (_Float16)r3.w;
    }

    float best_v[8];
    int   best_i[8];
    #pragma unroll
    for (int r = 0; r < 8; ++r) { best_v[r] = 3.0e38f; best_i[r] = 0; }

    union BB { v16h v; v8h h[2]; };
    BB b0[2], b1[2];

    for (int kh = 0; kh < 2; ++kh) {
        __syncthreads();  // wn ready (iter 0) / previous Bs consumers done (iter 1)

        // ---- stage w[f, :, kh*256 .. +256) transposed -> Bs[kcol][d] as f16.
        //      Thread: 8 d-rows x one k-quad; 4x ds_store_b128 per iteration ----
        #pragma unroll
        for (int it = 0; it < 2; ++it) {
            const int kg = tid & 63;                     // k-quad index
            const int d0 = it * 32 + (tid >> 6) * 8;     // 8 consecutive d rows
            float4 c[8];
            #pragma unroll
            for (int dd = 0; dd < 8; ++dd)
                c[dd] = *(const float4*)(wf + (d0 + dd) * K_ + kh * KH_COLS + kg * 4);
            #pragma unroll
            for (int i = 0; i < 4; ++i) {
                v8h hv;
                #pragma unroll
                for (int dd = 0; dd < 8; ++dd) {
                    float val = (i == 0) ? c[dd].x : (i == 1) ? c[dd].y
                              : (i == 2) ? c[dd].z : c[dd].w;
                    hv[dd] = (_Float16)val;
                }
                *(v8h*)(Bs + (kg * 4 + i) * BROW + d0) = hv;  // b128 store
            }
        }
        __syncthreads();

        // ---- 16 K-tiles of 16 columns, double-buffered B prefetch ----
        // preload tile 0
        {
            const _Float16* bp = Bs + m * BROW + hi * 8;
            b0[0].h[0] = *(const v8h*)(bp);
            b0[0].h[1] = *(const v8h*)(bp + 16);
            b1[0].h[0] = *(const v8h*)(bp + 32);
            b1[0].h[1] = *(const v8h*)(bp + 48);
        }
        #pragma unroll
        for (int kt = 0; kt < 16; ++kt) {
            const int cur = kt & 1, nxt = cur ^ 1;
            if (kt < 15) {   // prefetch next tile's B before consuming current
                const _Float16* bp = Bs + ((kt + 1) * 16 + m) * BROW + hi * 8;
                b0[nxt].h[0] = *(const v8h*)(bp);
                b0[nxt].h[1] = *(const v8h*)(bp + 16);
                b1[nxt].h[0] = *(const v8h*)(bp + 32);
                b1[nxt].h[1] = *(const v8h*)(bp + 48);
            }

            v8f acc = {};
            acc = __builtin_amdgcn_wmma_f32_16x16x32_f16(false, a0, false, b0[cur].v,
                                                         (short)0, acc, false, false);
            acc = __builtin_amdgcn_wmma_f32_16x16x32_f16(false, a1, false, b1[cur].v,
                                                         (short)0, acc, false, false);

            const int k  = kh * KH_COLS + kt * 16 + m;
            const float wk = wn[k];
            #pragma unroll
            for (int r = 0; r < 8; ++r) {
                float dist = wk - 2.0f * acc[r];   // ||x||^2 constant over k: dropped
                if (dist < best_v[r]) { best_v[r] = dist; best_i[r] = k; }
            }
        }
    }

    // ---- cross-lane argmin within 16-lane groups (first-index tie-break) ----
    int kstar[8];
    #pragma unroll
    for (int r = 0; r < 8; ++r) {
        float bv = best_v[r]; int bi = best_i[r];
        #pragma unroll
        for (int s = 1; s <= 8; s <<= 1) {
            float ov = __shfl_xor(bv, s, 32);
            int   oi = __shfl_xor(bi, s, 32);
            if (ov < bv || (ov == bv && oi < bi)) { bv = ov; bi = oi; }
        }
        kstar[r] = bi;   // row = r + 8*hi, replicated across the 16-lane group
    }

    // ---- distribute k*: lane handles (row = lane>>1, d-half = lane&1) ----
    const int row = lane >> 1;
    const int dh  = lane & 1;
    const int rr  = row & 7;
    const int src = (row < 8) ? 0 : 16;
    int kq = 0;
    #pragma unroll
    for (int r = 0; r < 8; ++r) {
        int v = __shfl(kstar[r], src, 32);
        if (rr == r) kq = v;
    }

    // ---- gather exact f32 code column, write output, accumulate loss ----
    const float* xg = xf + (size_t)(n0 + row) * D_ + dh * 32;
    float*       og = out + (size_t)f * N_ * D_ + (size_t)(n0 + row) * D_ + dh * 32;
    const float* wc = wf + kq;                 // w[f, d, kq], stride K_
    float ls = 0.f;
    #pragma unroll
    for (int q = 0; q < 8; ++q) {
        const int d = dh * 32 + q * 4;
        float4 xv = *(const float4*)(xg + q * 4);
        float4 qv;
        qv.x = wc[(d + 0) * K_];
        qv.y = wc[(d + 1) * K_];
        qv.z = wc[(d + 2) * K_];
        qv.w = wc[(d + 3) * K_];
        float e0 = qv.x - xv.x, e1 = qv.y - xv.y;
        float e2 = qv.z - xv.z, e3 = qv.w - xv.w;
        ls += e0 * e0 + e1 * e1 + e2 * e2 + e3 * e3;
        *(float4*)(og + q * 4) = qv;
    }

    // ---- deterministic block loss reduction (no atomics) ----
    #pragma unroll
    for (int s = 16; s >= 1; s >>= 1) ls += __shfl_xor(ls, s, 32);
    if (lane == 0) wls[wv] = ls;
    __syncthreads();
    if (tid == 0) {
        float t = 0.f;
        #pragma unroll
        for (int i = 0; i < 8; ++i) t += wls[i];
        partial[bid] = t;
    }
}

// ---------------------------------------------------------------------------
// Kernel 3: deterministic final loss reduction: 1024 partials -> scalar.
// ---------------------------------------------------------------------------
__global__ __launch_bounds__(32) void vq_loss_final(const float* __restrict__ partial,
                                                    float* __restrict__ loss)
{
    const int lane = threadIdx.x;
    float s = 0.f;
    for (int i = lane; i < 1024; i += 32) s += partial[i];
    #pragma unroll
    for (int mm = 16; mm >= 1; mm >>= 1) s += __shfl_xor(s, mm, 32);
    if (lane == 0)
        *loss = s * (1.25f / (float)((size_t)F_ * N_ * D_));  // q_loss + 0.25*e_loss
}

extern "C" void kernel_launch(void* const* d_in, const int* in_sizes, int n_in,
                              void* d_out, int out_size, void* d_ws, size_t ws_size,
                              hipStream_t stream) {
    (void)in_sizes; (void)n_in; (void)out_size; (void)ws_size;
    const float* x = (const float*)d_in[0];   // [F, N, D] f32
    const float* w = (const float*)d_in[1];   // [F, D, K] f32
    float* out     = (float*)d_out;           // [F, N, D] f32 ++ [1] loss
    float* partial = (float*)d_ws;            // [1024] block partials
    float* wnorm_g = (float*)d_ws + 1024;     // [F*K] = 8192 code norms

    vq_wnorm<<<dim3((F_ * K_) / 256), dim3(256), 0, stream>>>(w, wnorm_g);
    vq_main<<<dim3(1024), dim3(256), 0, stream>>>(x, w, wnorm_g, out, partial);
    vq_loss_final<<<dim3(1), dim3(32), 0, stream>>>(partial,
                                                    out + (size_t)F_ * N_ * D_);
}
